// GMTransv1_60378650247696
// MI455X (gfx1250) — matrix operand order repaired and checked by reading, not compiled, not run on hardware
//
#include <hip/hip_runtime.h>
#include <math.h>

// ---------------- constants ----------------
#define B_     2
#define C_     128
#define H_     192
#define W_     192
#define NH_    4
#define WSZ    8
#define NTOK   64
#define SHIFT_ 4
#define HID_   512
#define NWH    (H_ / WSZ)           // 24
#define NWW    (W_ / WSZ)           // 24
#define NWIN_  (NWH * NWW)          // 576
#define NBLK   (B_ * NWIN_)         // 1152
#define SCALE_ 0.17677669529663687f // 32^-0.5
#define EPS_   1e-5f

typedef __bf16 bf16;
typedef __attribute__((ext_vector_type(16))) __bf16 bf16x16;
typedef __attribute__((ext_vector_type(8)))  __bf16 bf16x8;
typedef __attribute__((ext_vector_type(4)))  __bf16 bf16x4;
typedef __attribute__((ext_vector_type(8)))  float  f32x8;

union FragU { bf16x16 v; bf16x8 h[2]; };

// ---------------- CDNA5 async global->LDS copy (16B per lane) ----------------
__device__ __forceinline__ void async_cp16(const bf16* __restrict__ gsrc,
                                           bf16* __restrict__ ldst) {
  // low 32 bits of a flat LDS pointer are the LDS byte offset (ISA 10.2)
  unsigned loff = (unsigned)(size_t)(void*)ldst;
  asm volatile("global_load_async_to_lds_b128 %0, %1, off"
               :
               : "v"(loff), "v"(gsrc)
               : "memory");
}
__device__ __forceinline__ void async_wait0() {
  asm volatile("s_wait_asynccnt 0" ::: "memory");
}

// ---------------- WMMA fragment helpers (CDNA5 wave32 layouts) ----------------
__device__ __forceinline__ bf16x16 load_a_vec(const bf16* __restrict__ A, int lda,
                                              int m0, int k0, int lane) {
  const bf16* p = A + (m0 + (lane & 15)) * lda + k0 + ((lane >> 4) << 3);
  FragU u;
  u.h[0] = *(const bf16x8*)p;
  u.h[1] = *(const bf16x8*)(p + 16);
  return u.v;
}

__device__ __forceinline__ bf16x16 load_bt_vec(const bf16* __restrict__ Bt, int ldbt,
                                               int n0, int k0, int lane) {
  const bf16* p = Bt + (n0 + (lane & 15)) * ldbt + k0 + ((lane >> 4) << 3);
  FragU u;
  u.h[0] = *(const bf16x8*)p;
  u.h[1] = *(const bf16x8*)(p + 16);
  return u.v;
}

__device__ __forceinline__ f32x8 wmma_bf16(bf16x16 a, bf16x16 b, f32x8 c) {
  return __builtin_amdgcn_wmma_f32_16x16x32_bf16(false, a, false, b, (short)0, c,
                                                 false, false);
}

__device__ __forceinline__ int band_(int v) {
  return (v < (H_ - WSZ)) ? 0 : ((v < (H_ - SHIFT_)) ? 1 : 2);
}

__device__ __forceinline__ int tok_base(int b, int h0, int w0, int t, int shift) {
  int hh = h0 + (t >> 3) + shift; if (hh >= H_) hh -= H_;
  int ww = w0 + (t & 7) + shift;  if (ww >= W_) ww -= W_;
  return ((b * H_ + hh) * W_ + ww) * C_;
}

// stage 64x128 f32 window tile -> LDS bf16 (needs convert; regular loads)
__device__ __forceinline__ void load_tile_f32(const float* __restrict__ src,
                                              bf16* __restrict__ dst, int b, int h0,
                                              int w0, int shift, int tid) {
  for (int i = tid; i < 2048; i += 256) {
    const int t = i >> 5, c4 = (i & 31) << 2;
    const float4 f = *(const float4*)(src + tok_base(b, h0, w0, t, shift) + c4);
    bf16x4 o = {(bf16)f.x, (bf16)f.y, (bf16)f.z, (bf16)f.w};
    *(bf16x4*)(dst + t * 128 + c4) = o;
  }
}

// stage 64x128 bf16 window tile -> LDS via async DMA path
__device__ __forceinline__ void load_tile_bf16_async(const bf16* __restrict__ src,
                                                     bf16* __restrict__ dst, int b,
                                                     int h0, int w0, int shift,
                                                     int tid) {
  for (int i = tid; i < 1024; i += 256) {
    const int t = i >> 4, c8 = (i & 15) << 3;
    async_cp16(src + tok_base(b, h0, w0, t, shift) + c8, dst + t * 128 + c8);
  }
  async_wait0();
}

// ---------------- prep kernels ----------------
__global__ __launch_bounds__(256) void tconv_kernel(const float* __restrict__ s,
                                                    bf16* __restrict__ d, int K, int N,
                                                    int total) {
  int i = blockIdx.x * 256 + threadIdx.x;
  if (i >= total) return;
  const int kn = K * N;
  const int dd = i / kn, r = i % kn;
  const int k = r / N, n = r % N;
  d[dd * kn + n * K + k] = (bf16)s[i];
}

__global__ __launch_bounds__(256) void convprep_kernel(const float* __restrict__ w,
                                                       bf16* __restrict__ o, int total) {
  int i = blockIdx.x * 256 + threadIdx.x;
  if (i >= total) return;
  const int dco = i / 2304, r = i % 2304;
  const int ci = r / 9, kp = r % 9;
  o[dco * 2304 + kp * 256 + ci] = (bf16)w[i];
}

__global__ __launch_bounds__(256) void fusepm_kernel(const float* __restrict__ pW,
                                                     const float* __restrict__ pb,
                                                     const float* __restrict__ mW,
                                                     bf16* __restrict__ w2t,
                                                     float* __restrict__ b2) {
  int i = blockIdx.x * 256 + threadIdx.x;
  if (i >= 2 * (16384 + 128)) return;
  const int d = i / (16384 + 128), r = i % (16384 + 128);
  const float* mWd = mW + d * 16384;
  if (r < 16384) {
    const int k = r >> 7, n = r & 127;
    const float* pWr = pW + d * 16384 + k * 128;
    float s = 0.0f;
    for (int j = 0; j < 128; ++j) s += pWr[j] * mWd[j * 128 + n];
    w2t[d * 16384 + n * 128 + k] = (bf16)s;
  } else {
    const int n = r - 16384;
    const float* pbd = pb + d * 128;
    float s = 0.0f;
    for (int j = 0; j < 128; ++j) s += pbd[j] * mWd[j * 128 + n];
    b2[d * 128 + n] = s;
  }
}

__global__ __launch_bounds__(256) void biasprep_kernel(const float* __restrict__ relb,
                                                       float* __restrict__ biasT) {
  int i = blockIdx.x * 256 + threadIdx.x;
  if (i >= 2 * NH_ * 4096) return;
  const int d = i >> 14, r = i & 16383;
  const int h = r >> 12, m = (r >> 6) & 63, n = r & 63;
  const int ridx = ((m >> 3) - (n >> 3) + 7) * 15 + ((m & 7) - (n & 7) + 7);
  biasT[i] = relb[d * 900 + ridx * NH_ + h];
}

__global__ __launch_bounds__(256) void nchw2bhwc_kernel(const float* __restrict__ in,
                                                        float* __restrict__ out) {
  int i = blockIdx.x * 256 + threadIdx.x;
  if (i >= B_ * C_ * H_ * W_) return;
  int w = i % W_, t = i / W_;
  int h = t % H_; t /= H_;
  int c = t % C_; int b = t / C_;
  out[((b * H_ + h) * W_ + w) * C_ + c] = in[i];
}

__global__ __launch_bounds__(256) void nchw2bhwc_bf16_kernel(const float* __restrict__ in,
                                                             bf16* __restrict__ out) {
  int i = blockIdx.x * 256 + threadIdx.x;
  if (i >= B_ * C_ * H_ * W_) return;
  int w = i % W_, t = i / W_;
  int h = t % H_; t /= H_;
  int c = t % C_; int b = t / C_;
  out[((b * H_ + h) * W_ + w) * C_ + c] = (bf16)in[i];
}

__global__ __launch_bounds__(256) void bhwc2nchw_kernel(const float* __restrict__ in,
                                                        float* __restrict__ out) {
  int i = blockIdx.x * 256 + threadIdx.x;
  if (i >= B_ * C_ * H_ * W_) return;
  int w = i % W_, t = i / W_;
  int h = t % H_; t /= H_;
  int c = t % C_; int b = t / C_;
  out[i] = in[((b * H_ + h) * W_ + w) * C_ + c];
}

// ---------------- fused window attention + FFN block ----------------
// per window: QKV, 4-head attention, fused proj, LN+residual, full MLP, LN+residual.
// x1 never leaves LDS.
__global__ __launch_bounds__(256) void block_kernel(
    const float* __restrict__ xh,  // [B,H,W,C] f32 trunk
    const bf16* __restrict__ fh,   // [B,H,W,C] bf16 feat
    const bf16* __restrict__ wqt, const float* __restrict__ qb,
    const bf16* __restrict__ wkvt, const float* __restrict__ kvb,
    const bf16* __restrict__ w2t, const float* __restrict__ b2,
    const float* __restrict__ biasT,
    const float* __restrict__ n1g, const float* __restrict__ n1b,
    const bf16* __restrict__ wf1t, const float* __restrict__ f1b,
    const bf16* __restrict__ wf2t, const float* __restrict__ f2b,
    const float* __restrict__ n2g, const float* __restrict__ n2b,
    int shift,
    bf16* __restrict__ outb) {
  __shared__ __align__(16) char smem[104 * 1024];
  bf16*  sX  = (bf16*)smem;                  // R0 staging
  bf16*  sQ  = (bf16*)(smem + 16 * 1024);    // R1
  bf16*  sK  = (bf16*)(smem + 32 * 1024);    // R2
  bf16*  sVt = (bf16*)(smem + 48 * 1024);    // R3
  float* sS  = (float*)(smem + 64 * 1024);   // R4
  bf16*  sO  = (bf16*)(smem + 80 * 1024);    // R5
  bf16*  sP  = (bf16*)(smem + 96 * 1024);    // R6
  float* sZ  = (float*)smem;                 // R0+R1 (Z, later FFN out)
  float* x1f = (float*)(smem + 32 * 1024);   // R2+R3 (x1 f32)
  bf16*  x1b = (bf16*)(smem + 64 * 1024);    // R4 (x1 bf16)
  bf16*  sH  = (bf16*)(smem + 80 * 1024);    // R5 (FFN hidden chunk)

  const int tid  = threadIdx.x;
  const int lane = tid & 31;
  const int wave = tid >> 5;
  const int hv   = lane >> 4;
  const int m0   = (wave & 3) << 4;
  const int nb   = (wave >> 2) << 4;

  const int b   = blockIdx.x / NWIN_;
  const int win = blockIdx.x % NWIN_;
  const int h0  = (win / NWW) * WSZ;
  const int w0  = (win % NWW) * WSZ;

  if (tid == 0) {  // weights live in the 192MB L2
    __builtin_prefetch(wqt, 0, 3);
    __builtin_prefetch(wkvt, 0, 3);
    __builtin_prefetch(w2t, 0, 3);
    __builtin_prefetch(wf1t, 0, 3);
    __builtin_prefetch(wf2t, 0, 3);
  }

  // ---- 1. feat window (async DMA, raw bf16) ----
  load_tile_bf16_async(fh, sX, b, h0, w0, shift, tid);
  __syncthreads();

  // ---- 2. K,V = fw @ kvW + kvb ----
  {
    bf16x16 a[4];
#pragma unroll
    for (int k = 0; k < 4; ++k) a[k] = load_a_vec(sX, 128, m0, k * 32, lane);
#pragma unroll
    for (int i = 0; i < 8; ++i) {
      const int n0 = nb + 32 * i;
      f32x8 acc = {};
#pragma unroll
      for (int k = 0; k < 4; ++k)
        acc = wmma_bf16(a[k], load_bt_vec(wkvt, 128, n0, k * 32, lane), acc);
      const int n = n0 + (lane & 15);
      const float bias = kvb[n];
      if (n < 128) {
#pragma unroll
        for (int v = 0; v < 8; ++v)
          sK[(m0 + v + (hv << 3)) * 128 + n] = (bf16)(acc[v] + bias);
      } else {
        bf16x8 pk;
#pragma unroll
        for (int v = 0; v < 8; ++v) pk[v] = (bf16)(acc[v] + bias);
        *(bf16x8*)(sVt + (n - 128) * 64 + m0 + (hv << 3)) = pk;
      }
    }
  }
  __syncthreads();

  // ---- 3. x window (f32 -> bf16) ----
  load_tile_f32(xh, sX, b, h0, w0, shift, tid);
  __syncthreads();

  // ---- 4. Q = (xw @ qW + qb) * scale ----
  {
    bf16x16 a[4];
#pragma unroll
    for (int k = 0; k < 4; ++k) a[k] = load_a_vec(sX, 128, m0, k * 32, lane);
#pragma unroll
    for (int i = 0; i < 4; ++i) {
      const int n0 = nb + 32 * i;
      f32x8 acc = {};
#pragma unroll
      for (int k = 0; k < 4; ++k)
        acc = wmma_bf16(a[k], load_bt_vec(wqt, 128, n0, k * 32, lane), acc);
      const int n = n0 + (lane & 15);
      const float bq = qb[n];
#pragma unroll
      for (int v = 0; v < 8; ++v)
        sQ[(m0 + v + (hv << 3)) * 128 + n] = (bf16)((acc[v] + bq) * SCALE_);
    }
  }
  __syncthreads();

  // ---- 5. per-head attention ----
  for (int h = 0; h < NH_; ++h) {
    {
      bf16x16 a = load_a_vec(sQ, 128, m0, h * 32, lane);
#pragma unroll
      for (int i = 0; i < 2; ++i) {
        const int n0 = nb + 32 * i;
        f32x8 z = {};
        f32x8 acc = wmma_bf16(a, load_bt_vec(sK + h * 32, 128, n0, 0, lane), z);
        const int n = n0 + (lane & 15);
#pragma unroll
        for (int v = 0; v < 8; ++v) sS[(m0 + v + (hv << 3)) * 64 + n] = acc[v];
      }
    }
    __syncthreads();
    {  // softmax (4 threads/row, bias+mask fused)
      const int r = tid >> 2, q = tid & 3;
      const float* srow = sS + r * 64 + q * 16;
      const float* brow = biasT + h * 4096 + r * 64 + q * 16;
      float vals[16];
#pragma unroll
      for (int c4 = 0; c4 < 4; ++c4) {
        float4 sv = *(const float4*)(srow + c4 * 4);
        float4 bv = *(const float4*)(brow + c4 * 4);
        vals[c4 * 4 + 0] = sv.x + bv.x;
        vals[c4 * 4 + 1] = sv.y + bv.y;
        vals[c4 * 4 + 2] = sv.z + bv.z;
        vals[c4 * 4 + 3] = sv.w + bv.w;
      }
      if (shift) {
        const int lm = band_(h0 + (r >> 3)) * 3 + band_(w0 + (r & 7));
#pragma unroll
        for (int c = 0; c < 16; ++c) {
          const int j = q * 16 + c;
          const int ln = band_(h0 + (j >> 3)) * 3 + band_(w0 + (j & 7));
          if (ln != lm) vals[c] -= 100.0f;
        }
      }
      float mx = vals[0];
#pragma unroll
      for (int c = 1; c < 16; ++c) mx = fmaxf(mx, vals[c]);
      mx = fmaxf(mx, __shfl_xor(mx, 1, 4));
      mx = fmaxf(mx, __shfl_xor(mx, 2, 4));
      float sum = 0.0f;
#pragma unroll
      for (int c = 0; c < 16; ++c) {
        vals[c] = __expf(vals[c] - mx);
        sum += vals[c];
      }
      sum += __shfl_xor(sum, 1, 4);
      sum += __shfl_xor(sum, 2, 4);
      const float inv = 1.0f / sum;
      bf16* prow = sP + r * 64 + q * 16;
#pragma unroll
      for (int c4 = 0; c4 < 4; ++c4) {
        bf16x4 o = {(bf16)(vals[c4 * 4 + 0] * inv), (bf16)(vals[c4 * 4 + 1] * inv),
                    (bf16)(vals[c4 * 4 + 2] * inv), (bf16)(vals[c4 * 4 + 3] * inv)};
        *(bf16x4*)(prow + c4 * 4) = o;
      }
    }
    __syncthreads();
    {  // O_h = P @ V_h
      f32x8 acc = {};
#pragma unroll
      for (int k = 0; k < 2; ++k)
        acc = wmma_bf16(load_a_vec(sP, 64, m0, k * 32, lane),
                        load_bt_vec(sVt + h * 32 * 64, 64, nb, k * 32, lane), acc);
      const int n = nb + (lane & 15);
#pragma unroll
      for (int v = 0; v < 8; ++v)
        sO[(m0 + v + (hv << 3)) * 128 + h * 32 + n] = (bf16)acc[v];
    }
    __syncthreads();
  }

  // ---- 6. Z = AO @ (pW@mW) + pb@mW ----
  {
    bf16x16 a[4];
#pragma unroll
    for (int k = 0; k < 4; ++k) a[k] = load_a_vec(sO, 128, m0, k * 32, lane);
#pragma unroll
    for (int i = 0; i < 4; ++i) {
      const int n0 = nb + 32 * i;
      f32x8 acc = {};
#pragma unroll
      for (int k = 0; k < 4; ++k)
        acc = wmma_bf16(a[k], load_bt_vec(w2t, 128, n0, k * 32, lane), acc);
      const int n = n0 + (lane & 15);
      const float bb = b2[n];
#pragma unroll
      for (int v = 0; v < 8; ++v) sZ[(m0 + v + (hv << 3)) * 128 + n] = acc[v] + bb;
    }
  }
  __syncthreads();

  // ---- 7. LN1 + shortcut residual -> x1 kept in LDS (f32 + bf16) ----
  {
    const int r = tid >> 2, q = tid & 3;
    const float4* z4 = (const float4*)(sZ + r * 128 + q * 32);
    float s = 0.0f;
#pragma unroll
    for (int c = 0; c < 8; ++c) {
      float4 t = z4[c];
      s += t.x + t.y + t.z + t.w;
    }
    s += __shfl_xor(s, 1, 4);
    s += __shfl_xor(s, 2, 4);
    const float mean = s * (1.0f / 128.0f);
    float v2 = 0.0f;
#pragma unroll
    for (int c = 0; c < 8; ++c) {
      float4 t = z4[c];
      float a0 = t.x - mean, a1 = t.y - mean, a2 = t.z - mean, a3 = t.w - mean;
      v2 += a0 * a0 + a1 * a1 + a2 * a2 + a3 * a3;
    }
    v2 += __shfl_xor(v2, 1, 4);
    v2 += __shfl_xor(v2, 2, 4);
    const float rs = rsqrtf(v2 * (1.0f / 128.0f) + EPS_);
    const int base = tok_base(b, h0, w0, r, shift) + q * 32;
    float* xf = x1f + r * 128 + q * 32;
    bf16*  xb = x1b + r * 128 + q * 32;
#pragma unroll
    for (int c = 0; c < 8; ++c) {
      float4 z = z4[c];
      float4 xv = *(const float4*)(xh + base + c * 4);
      float4 g = *(const float4*)(n1g + q * 32 + c * 4);
      float4 bb = *(const float4*)(n1b + q * 32 + c * 4);
      float4 o;
      o.x = xv.x + (z.x - mean) * rs * g.x + bb.x;
      o.y = xv.y + (z.y - mean) * rs * g.y + bb.y;
      o.z = xv.z + (z.z - mean) * rs * g.z + bb.z;
      o.w = xv.w + (z.w - mean) * rs * g.w + bb.w;
      *(float4*)(xf + c * 4) = o;
      bf16x4 ob = {(bf16)o.x, (bf16)o.y, (bf16)o.z, (bf16)o.w};
      *(bf16x4*)(xb + c * 4) = ob;
    }
  }
  __syncthreads();

  // ---- 8. FFN: h=gelu(x1@f1+b1); acc = h@f2 ----
  bf16x16 aA[4];
#pragma unroll
  for (int k = 0; k < 4; ++k) aA[k] = load_a_vec(x1b, 128, m0, k * 32, lane);
  f32x8 facc[4] = {{}, {}, {}, {}};
  for (int cc = 0; cc < 4; ++cc) {
#pragma unroll
    for (int i = 0; i < 4; ++i) {
      const int n0 = nb + 32 * i;
      f32x8 a = {};
#pragma unroll
      for (int k = 0; k < 4; ++k)
        a = wmma_bf16(aA[k], load_bt_vec(wf1t + cc * 16384, 128, n0, k * 32, lane), a);
      const int n = n0 + (lane & 15);
      const float bb = f1b[cc * 128 + n];
#pragma unroll
      for (int v = 0; v < 8; ++v) {
        float hx = a[v] + bb;
        float g = 0.5f * hx * (1.0f + erff(hx * 0.70710678118654752f));
        sH[(m0 + v + (hv << 3)) * 128 + n] = (bf16)g;
      }
    }
    __syncthreads();
    bf16x16 aH[4];
#pragma unroll
    for (int k = 0; k < 4; ++k) aH[k] = load_a_vec(sH, 128, m0, k * 32, lane);
#pragma unroll
    for (int i = 0; i < 4; ++i) {
      const int n0 = nb + 32 * i;
#pragma unroll
      for (int k = 0; k < 4; ++k)
        facc[i] = wmma_bf16(aH[k], load_bt_vec(wf2t, 512, n0, cc * 128 + k * 32, lane),
                            facc[i]);
    }
    __syncthreads();
  }
  // ---- 9. z2 = acc + f2b -> LDS ----
#pragma unroll
  for (int i = 0; i < 4; ++i) {
    const int n = nb + 32 * i + (lane & 15);
    const float bb = f2b[n];
#pragma unroll
    for (int v = 0; v < 8; ++v) sZ[(m0 + v + (hv << 3)) * 128 + n] = facc[i][v] + bb;
  }
  __syncthreads();

  // ---- 10. LN2 + residual(x1) -> branch output (bf16, global) ----
  {
    const int r = tid >> 2, q = tid & 3;
    const float4* z4 = (const float4*)(sZ + r * 128 + q * 32);
    float s = 0.0f;
#pragma unroll
    for (int c = 0; c < 8; ++c) {
      float4 t = z4[c];
      s += t.x + t.y + t.z + t.w;
    }
    s += __shfl_xor(s, 1, 4);
    s += __shfl_xor(s, 2, 4);
    const float mean = s * (1.0f / 128.0f);
    float v2 = 0.0f;
#pragma unroll
    for (int c = 0; c < 8; ++c) {
      float4 t = z4[c];
      float a0 = t.x - mean, a1 = t.y - mean, a2 = t.z - mean, a3 = t.w - mean;
      v2 += a0 * a0 + a1 * a1 + a2 * a2 + a3 * a3;
    }
    v2 += __shfl_xor(v2, 1, 4);
    v2 += __shfl_xor(v2, 2, 4);
    const float rs = rsqrtf(v2 * (1.0f / 128.0f) + EPS_);
    const float* xf = x1f + r * 128 + q * 32;
    const int base = tok_base(b, h0, w0, r, shift) + q * 32;
#pragma unroll
    for (int c = 0; c < 8; ++c) {
      float4 z = z4[c];
      float4 xv = *(const float4*)(xf + c * 4);
      float4 g = *(const float4*)(n2g + q * 32 + c * 4);
      float4 bb = *(const float4*)(n2b + q * 32 + c * 4);
      bf16x4 o = {(bf16)(xv.x + (z.x - mean) * rs * g.x + bb.x),
                  (bf16)(xv.y + (z.y - mean) * rs * g.y + bb.y),
                  (bf16)(xv.z + (z.z - mean) * rs * g.z + bb.z),
                  (bf16)(xv.w + (z.w - mean) * rs * g.w + bb.w)};
      *(bf16x4*)(outb + base + c * 4) = o;
    }
  }
}

// ---------------- implicit-GEMM 3x3 conv + PReLU (async double-buffered) ----------------
__global__ __launch_bounds__(256) void conv_kernel(
    const bf16* __restrict__ sa, const bf16* __restrict__ ta,
    const bf16* __restrict__ wc, const float* __restrict__ cb,
    const float* __restrict__ pa, float* __restrict__ outx) {
  __shared__ __align__(16) bf16 sA2[2][NTOK * 32];

  const int tid  = threadIdx.x;
  const int lane = tid & 31;
  const int wave = tid >> 5;
  const int hv   = lane >> 4;
  const int m0   = (wave & 3) << 4;
  const int nb   = (wave >> 2) << 4;

  const int b    = blockIdx.x / NWIN_;
  const int tile = blockIdx.x % NWIN_;
  const int h0   = (tile / NWW) * WSZ;
  const int w0   = (tile % NWW) * WSZ;

  const int p  = tid >> 2;
  const int c8 = (tid & 3) << 3;

  if (tid == 0) __builtin_prefetch(wc, 0, 3);

  auto stage_issue = [&](int s, bf16* sA) {
    const int kp = s >> 3;
    const int kh = kp / 3, kw = kp % 3;
    const int c0 = (s & 7) << 5;
    const int hh = h0 + (p >> 3) + kh - 1;
    const int ww = w0 + (p & 7) + kw - 1;
    bf16* dstp = sA + p * 32 + c8;
    if (hh >= 0 && hh < H_ && ww >= 0 && ww < W_) {
      const int cg = c0 + c8;
      const bf16* srcp = (cg < 128) ? sa : ta;
      async_cp16(srcp + ((b * H_ + hh) * W_ + ww) * 128 + (cg & 127), dstp);
    } else {
      bf16x8 zz = {};
      *(bf16x8*)dstp = zz;
    }
  };

  f32x8 acc[4] = {{}, {}, {}, {}};
  stage_issue(0, sA2[0]);
  for (int s = 0; s < 72; ++s) {
    async_wait0();
    __syncthreads();
    if (s < 71) stage_issue(s + 1, sA2[(s + 1) & 1]);  // overlaps WMMAs below
    const bf16* sA = sA2[s & 1];
    const int kp = s >> 3;
    const int c0 = (s & 7) << 5;
    const bf16* Bp = wc + kp * 256 + c0;
    bf16x16 a = load_a_vec(sA, 32, m0, 0, lane);
#pragma unroll
    for (int i = 0; i < 4; ++i)
      acc[i] = wmma_bf16(a, load_bt_vec(Bp, 2304, nb + 32 * i, 0, lane), acc[i]);
  }
#pragma unroll
  for (int i = 0; i < 4; ++i) {
    const int n = nb + 32 * i + (lane & 15);
    const float bias = cb[n], alpha = pa[n];
#pragma unroll
    for (int v = 0; v < 8; ++v) {
      const int pp = m0 + v + (hv << 3);
      float y = acc[i][v] + bias;
      y = (y >= 0.0f) ? y : alpha * y;
      outx[((b * H_ + h0 + (pp >> 3)) * W_ + w0 + (pp & 7)) * 128 + n] = y;
    }
  }
}

// ---------------- driver ----------------
extern "C" void kernel_launch(void* const* d_in, const int* in_sizes, int n_in,
                              void* d_out, int out_size, void* d_ws, size_t ws_size,
                              hipStream_t stream) {
  (void)in_sizes; (void)n_in; (void)out_size; (void)ws_size;
  const float* x    = (const float*)d_in[0];
  const float* src  = (const float*)d_in[1];
  const float* tgt  = (const float*)d_in[2];
  const float* qW   = (const float*)d_in[3];
  const float* qb   = (const float*)d_in[4];
  const float* kvW  = (const float*)d_in[5];
  const float* kvb  = (const float*)d_in[6];
  const float* pW   = (const float*)d_in[7];
  const float* pb   = (const float*)d_in[8];
  const float* relb = (const float*)d_in[9];
  const float* mW   = (const float*)d_in[10];
  const float* n1g  = (const float*)d_in[11];
  const float* n1b  = (const float*)d_in[12];
  const float* f1W  = (const float*)d_in[13];
  const float* f1b  = (const float*)d_in[14];
  const float* f2W  = (const float*)d_in[15];
  const float* f2b  = (const float*)d_in[16];
  const float* n2g  = (const float*)d_in[17];
  const float* n2b  = (const float*)d_in[18];
  const float* cW   = (const float*)d_in[19];
  const float* cb   = (const float*)d_in[20];
  const float* pa   = (const float*)d_in[21];

  size_t off = 0;
  auto take = [&](size_t bytes) -> char* {
    char* p = (char*)d_ws + off;
    off += (bytes + 255) & ~(size_t)255;
    return p;
  };
  const size_t NEL = (size_t)B_ * H_ * W_ * C_;
  float* bx    = (float*)take(NEL * 4);
  bf16* bs     = (bf16*)take(NEL * 2);
  bf16* bt     = (bf16*)take(NEL * 2);
  bf16* bsa    = (bf16*)take(NEL * 2);
  bf16* bta    = (bf16*)take(NEL * 2);
  bf16* wqt    = (bf16*)take((size_t)2 * 128 * 128 * 2);
  bf16* wkvt   = (bf16*)take((size_t)2 * 128 * 256 * 2);
  bf16* w2t    = (bf16*)take((size_t)2 * 128 * 128 * 2);
  float* b2    = (float*)take((size_t)2 * 128 * 4);
  bf16* wf1t   = (bf16*)take((size_t)2 * 128 * 512 * 2);
  bf16* wf2t   = (bf16*)take((size_t)2 * 512 * 128 * 2);
  bf16* wc     = (bf16*)take((size_t)2 * 9 * 256 * 128 * 2);
  float* biasT = (float*)take((size_t)2 * NH_ * 4096 * 4);

  auto tcv = [&](const float* s, bf16* d, int K, int N) {
    int n = 2 * K * N;
    tconv_kernel<<<(n + 255) / 256, 256, 0, stream>>>(s, d, K, N, n);
  };
  tcv(qW, wqt, 128, 128);
  tcv(kvW, wkvt, 128, 256);
  tcv(f1W, wf1t, 128, 512);
  tcv(f2W, wf2t, 512, 128);
  fusepm_kernel<<<(2 * 16512 + 255) / 256, 256, 0, stream>>>(pW, pb, mW, w2t, b2);
  convprep_kernel<<<(589824 + 255) / 256, 256, 0, stream>>>(cW, wc, 589824);
  biasprep_kernel<<<(32768 + 255) / 256, 256, 0, stream>>>(relb, biasT);

  const int tgrid = (int)((NEL + 255) / 256);
  nchw2bhwc_kernel<<<tgrid, 256, 0, stream>>>(x, bx);
  nchw2bhwc_bf16_kernel<<<tgrid, 256, 0, stream>>>(src, bs);
  nchw2bhwc_bf16_kernel<<<tgrid, 256, 0, stream>>>(tgt, bt);

  for (int d = 0; d < 2; ++d) {
    const int shift = d ? SHIFT_ : 0;
    for (int br = 0; br < 2; ++br) {
      const bf16* feat = br ? bt : bs;
      bf16* outb = br ? bta : bsa;
      block_kernel<<<NBLK, 256, 0, stream>>>(
          bx, feat,
          wqt + d * 16384, qb + d * 128,
          wkvt + d * 32768, kvb + d * 256,
          w2t + d * 16384, b2 + d * 128,
          biasT + d * NH_ * 4096,
          n1g + d * 128, n1b + d * 128,
          wf1t + d * 65536, f1b + d * 512,
          wf2t + d * 65536, f2b + d * 128,
          n2g + d * 128, n2b + d * 128,
          shift, outb);
    }
    conv_kernel<<<NBLK, 256, 0, stream>>>(
        bsa, bta, wc + (size_t)d * 294912, cb + d * 128, pa + d * 128, bx);
  }
  bhwc2nchw_kernel<<<tgrid, 256, 0, stream>>>(bx, (float*)d_out);
}
